// Classifier_90555090469072
// MI455X (gfx1250) — matrix operand order repaired and checked by reading, not compiled, not run on hardware
//
#include <hip/hip_runtime.h>
#include <hip/hip_bf16.h>

// ---------------- dims ----------------
#define BATCH 4096
#define HH    200
#define DD    400          // 2*H
#define GG    200
#define NF    11
#define NHOP  4

typedef __attribute__((ext_vector_type(16))) __bf16 v16bf;
typedef __attribute__((ext_vector_type(8)))  __bf16 v8bf;
typedef __attribute__((ext_vector_type(8)))  float  v8f;
typedef __attribute__((ext_vector_type(4)))  float  v4f;

__device__ inline float  bf2f(__bf16 x) { return (float)x; }
__device__ inline __bf16 f2bf(float x)  { return (__bf16)x; }
__device__ inline v8bf   zero8()        { v8bf z;
#pragma unroll
  for (int i = 0; i < 8; ++i) z[i] = (__bf16)0.0f; return z; }

// WMMA D = A(16x32 bf16) x B(32x16 bf16) + C(16x16 f32)
__device__ inline v8f wmma_bf16(v16bf a, v16bf b, v8f c) {
  return __builtin_amdgcn_wmma_f32_16x16x32_bf16(false, a, false, b, (short)0, c, false, false);
}

// A fragment from LDS tile At[r][32] (row-major): two contiguous 16B runs.
__device__ inline v16bf ldsA(const __bf16* At, int lane, int m0) {
  int r  = m0 + (lane & 15);
  int ko = (lane & 16) ? 16 : 0;
  const __bf16* p = At + r * 32 + ko;
  v8bf lo = *(const v8bf*)p;
  v8bf hi = *(const v8bf*)(p + 8);
  return __builtin_shufflevector(lo, hi, 0, 1, 2, 3, 4, 5, 6, 7, 8, 9, 10, 11, 12, 13, 14, 15);
}

// B fragment from LDS tile Bt[n][32] (N-major): one contiguous 32B run.
__device__ inline v16bf ldsBT(const __bf16* Bt, int lane, int n0) {
  int n  = n0 + (lane & 15);
  int ko = (lane & 16) ? 16 : 0;
  return *(const v16bf*)(Bt + n * 32 + ko);
}

// generic shared pointer -> 32-bit LDS byte offset (low 32 bits of the generic address)
__device__ inline unsigned lds_off(const void* p) {
  return (unsigned)(unsigned long long)p;
}

// async DMA: global (SADDR base + 32-bit VGPR offset) -> LDS, 16B per lane, ASYNCcnt-tracked
__device__ inline void async_g2l_b128(const void* gbase, unsigned gbyteoff, unsigned ldsbyte) {
  asm volatile("global_load_async_to_lds_b128 %0, %1, %2"
               :: "v"(ldsbyte), "v"(gbyteoff), "s"(gbase)
               : "memory");
}

__device__ inline void wait_async() {
#if __has_builtin(__builtin_amdgcn_s_wait_asynccnt)
  __builtin_amdgcn_s_wait_asynccnt(0);
#else
  asm volatile("s_wait_asynccnt 0x0" ::: "memory");
#endif
}

// stage transposed weights WT[n][Kpad] -> Bt[n][32] via async LDS DMA (8-bf16 = 16B chunks)
__device__ inline void stageBT(const __bf16* __restrict__ WT, __bf16* Bt,
                               int ncols, int Kpad, int kk0, int tid) {
  for (int idx = tid; idx < ncols * 4; idx += 256) {
    int n = idx >> 2, c8 = (idx & 3) << 3;
    async_g2l_b128(WT, (unsigned)((n * Kpad + kk0 + c8) * 2),
                   lds_off(Bt + n * 32 + c8));
  }
}

// ---------------- prep ----------------
__global__ void pack_bf16(const float* __restrict__ src, __bf16* __restrict__ dst, int total) {
  for (int idx = blockIdx.x * blockDim.x + threadIdx.x; idx < total;
       idx += gridDim.x * blockDim.x)
    dst[idx] = f2bf(src[idx]);
}

// fp32 (K,N) -> transposed padded bf16 [dn0+n][Kpad]
__global__ void packT_bf16(const float* __restrict__ src, __bf16* __restrict__ dst,
                           int srows, int scols, int Kpad, int dn0, int dnw) {
  int total = dnw * Kpad;
  for (int idx = blockIdx.x * blockDim.x + threadIdx.x; idx < total;
       idx += gridDim.x * blockDim.x) {
    int n = idx / Kpad, k = idx - n * Kpad;
    float v = (k < srows && n < scols) ? src[k * scols + n] : 0.f;
    dst[(dn0 + n) * Kpad + k] = f2bf(v);
  }
}

__global__ void init_mem_kernel(float* mem, __bf16* memb) {
  int i = blockIdx.x * blockDim.x + threadIdx.x;
  if (i < BATCH * GG)  mem[i]  = 0.f;
  if (i < BATCH * 208) memb[i] = f2bf(0.f);
}

// ---------------- hop kernel A: mib = bf16(mem @ Wm + bm) ----------------
__global__ void mi_kernel(const __bf16* __restrict__ memb, const __bf16* __restrict__ WmT,
                          const float* __restrict__ bm, __bf16* __restrict__ mib) {
  __shared__ alignas(64) __bf16 At[16 * 32];
  __shared__ alignas(64) __bf16 Bt[512 * 32];
  int tid = threadIdx.x, lane = tid & 31, wave = tid >> 5;
  int b0 = blockIdx.x * 16;
  v8f acc[4] = {};
  for (int step = 0; step < 7; ++step) {
    int kk0 = step * 32;
    stageBT(WmT, Bt, 512, 224, kk0, tid);
    for (int idx = tid; idx < 16 * 4; idx += 256) {
      int r = idx >> 2, c8 = (idx & 3) << 3, k = kk0 + c8;
      v8bf v = (k < 208) ? *(const v8bf*)(memb + (b0 + r) * 208 + k) : zero8();
      *(v8bf*)(At + r * 32 + c8) = v;
    }
    wait_async();
    __syncthreads();
    v16bf a = ldsA(At, lane, 0);
#pragma unroll
    for (int ti = 0; ti < 4; ++ti)
      acc[ti] = wmma_bf16(a, ldsBT(Bt, lane, (wave + ti * 8) * 16), acc[ti]);
    __syncthreads();
  }
#pragma unroll
  for (int ti = 0; ti < 4; ++ti) {
    int n = (wave + ti * 8) * 16 + (lane & 15);
    if (n < DD) {
#pragma unroll
      for (int v = 0; v < 8; ++v) {
        int b = b0 + v + ((lane & 16) ? 8 : 0);
        mib[b * DD + n] = f2bf(acc[ti][v] + bm[n]);
      }
    }
  }
}

// -------- hop kernel B: fused features -> relu(ref@W1+b1)@W2 -> logits (B,NF) --------
__global__ void logits_kernel(const __bf16* __restrict__ qb, const __bf16* __restrict__ cb,
                              const __bf16* __restrict__ kb, const __bf16* __restrict__ mib,
                              const __bf16* __restrict__ W1T, const float* __restrict__ b1,
                              const float* __restrict__ W2, float* __restrict__ logits) {
  __shared__ alignas(64) __bf16 At[48 * 32];
  __shared__ alignas(64) __bf16 Bt[208 * 32];
  __shared__ float logsum[48];
  int tid = threadIdx.x, lane = tid & 31, wave = tid >> 5;
  int bb0 = blockIdx.x * 4;
  v8f acc[5] = {};
  for (int step = 0; step < 75; ++step) {
    int kk0 = step * 32;
    stageBT(W1T, Bt, 208, 2400, kk0, tid);
    // feature A tile, 8-wide (feature-block boundary 400 is 8-aligned)
    for (int idx = tid; idx < 48 * 4; idx += 256) {
      int r = idx >> 2, c8 = (idx & 3) << 3;
      int bi = r / 12, f = r - bi * 12;
      int kk = kk0 + c8;
      int bl = kk / 400, d = kk - bl * 400;
      v8bf out;
      if (f < NF) {
        int b = bb0 + bi;
        v8bf k8 = *(const v8bf*)(kb + (b * NF + f) * DD + d);
        const __bf16* xsrc = (bl == 0 || bl == 3) ? qb : (bl == 1 || bl == 4) ? cb : mib;
        v8bf x8 = *(const v8bf*)(xsrc + b * DD + d);
        if (bl < 3) {
#pragma unroll
          for (int i = 0; i < 8; ++i) out[i] = f2bf(bf2f(x8[i]) * bf2f(k8[i]));
        } else {
#pragma unroll
          for (int i = 0; i < 8; ++i) out[i] = f2bf(fabsf(bf2f(x8[i]) - bf2f(k8[i])));
        }
      } else {
        out = zero8();
      }
      *(v8bf*)(At + r * 32 + c8) = out;
    }
    wait_async();
    __syncthreads();
#pragma unroll
    for (int ti = 0; ti < 5; ++ti) {
      int t = wave + ti * 8;
      int tt = (t < 39) ? t : 0;          // wave 7's 5th tile is a dummy recompute of tile 0
      int mt = tt / 13, nt = tt - mt * 13;
      v16bf a = ldsA(At, lane, mt * 16);
      acc[ti] = wmma_bf16(a, ldsBT(Bt, lane, nt * 16), acc[ti]);
    }
    __syncthreads();
  }
  if (tid < 48) logsum[tid] = 0.f;
  __syncthreads();
#pragma unroll
  for (int ti = 0; ti < 5; ++ti) {
    int t = wave + ti * 8;
    if (t < 39) {
      int mt = t / 13, nt = t - mt * 13;
      int n = nt * 16 + (lane & 15);
      if (n < HH) {
        float w2 = W2[n], bb = b1[n];
#pragma unroll
        for (int v = 0; v < 8; ++v) {
          int r = mt * 16 + v + ((lane & 16) ? 8 : 0);
          float h = acc[ti][v] + bb;
          h = h > 0.f ? h : 0.f;
          atomicAdd(&logsum[r], h * w2);
        }
      }
    }
  }
  __syncthreads();
  if (tid < 48) {
    int bi = tid / 12, f = tid - bi * 12;
    if (f < NF) logits[(bb0 + bi) * NF + f] = logsum[tid];
  }
}

// -------- hop kernel C: softmax over NF, retrieved = sum(att*k) --------
__global__ void softmax_retrieve_kernel(const float* __restrict__ logits,
                                        const __bf16* __restrict__ kb,
                                        float* __restrict__ att_out,
                                        __bf16* __restrict__ retb) {
  int b = blockIdx.x, tid = threadIdx.x;
  __shared__ float att[NF];
  if (tid == 0) {
    float l[NF], m = -1e30f;
    for (int f = 0; f < NF; ++f) { l[f] = logits[b * NF + f]; m = fmaxf(m, l[f]); }
    float s = 0.f;
    for (int f = 0; f < NF; ++f) { l[f] = expf(l[f] - m); s += l[f]; }
    float inv = 1.f / s;
    for (int f = 0; f < NF; ++f) { att[f] = l[f] * inv; att_out[b * NF + f] = att[f]; }
  }
  __syncthreads();
  for (int d = tid; d < DD; d += 64) {
    float s = 0.f;
#pragma unroll
    for (int f = 0; f < NF; ++f) s += att[f] * bf2f(kb[(b * NF + f) * DD + d]);
    retb[b * DD + d] = f2bf(s);
  }
}

// -------- hop kernel D: [Az|Ar|Ahw] = mri@[Wz|Wr|Wh] + mem@[Uz|Ur|0] --------
__global__ void gates_kernel(const __bf16* __restrict__ qb, const __bf16* __restrict__ cb,
                             const __bf16* __restrict__ retb, const __bf16* __restrict__ WzrhT,
                             const __bf16* __restrict__ memb, const __bf16* __restrict__ UzrT,
                             const float* __restrict__ bz, const float* __restrict__ buz,
                             const float* __restrict__ br, const float* __restrict__ bur,
                             float* __restrict__ zbuf, float* __restrict__ rbuf,
                             float* __restrict__ ahw) {
  __shared__ alignas(64) __bf16 At[16 * 32];
  __shared__ alignas(64) __bf16 Bt[640 * 32];
  int tid = threadIdx.x, lane = tid & 31, wave = tid >> 5;
  int b0 = blockIdx.x * 16;
  v8f acc[5] = {};
  for (int step = 0; step < 88; ++step) {
    int kk0 = step * 32;
    stageBT(WzrhT, Bt, 640, 2816, kk0, tid);
    for (int idx = tid; idx < 16 * 4; idx += 256) {
      int r = idx >> 2, c8 = (idx & 3) << 3, kk = kk0 + c8;
      v8bf out;
      if (kk < 2800) {
        int bl = kk / 400, d = kk - bl * 400;
        int b = b0 + r;
        v8bf q8 = *(const v8bf*)(qb + b * DD + d);
        v8bf c8v = *(const v8bf*)(cb + b * DD + d);
        v8bf r8 = *(const v8bf*)(retb + b * DD + d);
#pragma unroll
        for (int i = 0; i < 8; ++i) {
          float q = bf2f(q8[i]), cv = bf2f(c8v[i]), rv = bf2f(r8[i]);
          float val;
          switch (bl) {
            case 0: val = q; break;
            case 1: val = cv; break;
            case 2: val = rv; break;
            case 3: val = q * rv; break;
            case 4: val = cv * rv; break;
            case 5: val = fabsf(q - rv); break;
            default: val = fabsf(cv - rv); break;
          }
          out[i] = f2bf(val);
        }
      } else {
        out = zero8();
      }
      *(v8bf*)(At + r * 32 + c8) = out;
    }
    wait_async();
    __syncthreads();
    v16bf a = ldsA(At, lane, 0);
#pragma unroll
    for (int ti = 0; ti < 5; ++ti)
      acc[ti] = wmma_bf16(a, ldsBT(Bt, lane, (wave + ti * 8) * 16), acc[ti]);
    __syncthreads();
  }
  // + mem @ [Uz|Ur|0...] (UzrT zero-padded to 640 cols -> same accumulators)
  for (int step = 0; step < 7; ++step) {
    int kk0 = step * 32;
    stageBT(UzrT, Bt, 640, 224, kk0, tid);
    for (int idx = tid; idx < 16 * 4; idx += 256) {
      int r = idx >> 2, c8 = (idx & 3) << 3, k = kk0 + c8;
      v8bf v = (k < 208) ? *(const v8bf*)(memb + (b0 + r) * 208 + k) : zero8();
      *(v8bf*)(At + r * 32 + c8) = v;
    }
    wait_async();
    __syncthreads();
    v16bf a = ldsA(At, lane, 0);
#pragma unroll
    for (int ti = 0; ti < 5; ++ti)
      acc[ti] = wmma_bf16(a, ldsBT(Bt, lane, (wave + ti * 8) * 16), acc[ti]);
    __syncthreads();
  }
#pragma unroll
  for (int ti = 0; ti < 5; ++ti) {
    int col = (wave + ti * 8) * 16 + (lane & 15);
    int g = col / 208, nn = col - g * 208;
    if (g < 3 && nn < GG) {
#pragma unroll
      for (int v = 0; v < 8; ++v) {
        int b = b0 + v + ((lane & 16) ? 8 : 0);
        float x = acc[ti][v];
        if (g == 0)      zbuf[b * GG + nn] = 1.f / (1.f + expf(-(x + bz[nn] + buz[nn])));
        else if (g == 1) rbuf[b * GG + nn] = 1.f / (1.f + expf(-(x + br[nn] + bur[nn])));
        else             ahw[b * GG + nn]  = x;
      }
    }
  }
}

// -------- hop kernel E: h = tanh(Ahw + (r*mem)@Uh + bh+buh); mem = (1-z)mem + z*h --------
__global__ void update_kernel(const float* __restrict__ rbuf, const float* __restrict__ zbuf,
                              const float* __restrict__ ahw, const __bf16* __restrict__ UhT,
                              const float* __restrict__ bh, const float* __restrict__ buh,
                              float* __restrict__ mem, __bf16* __restrict__ memb) {
  __shared__ alignas(64) __bf16 At[16 * 32];
  __shared__ alignas(64) __bf16 Bt[256 * 32];
  int tid = threadIdx.x, lane = tid & 31, wave = tid >> 5;
  int b0 = blockIdx.x * 16;
  v8f acc[2] = {};
  for (int step = 0; step < 7; ++step) {
    int kk0 = step * 32;
    stageBT(UhT, Bt, 256, 224, kk0, tid);
    for (int idx = tid; idx < 16 * 4; idx += 256) {
      int r = idx >> 2, c8 = (idx & 3) << 3, k = kk0 + c8;
      v8bf out;
      if (k < GG) {
        const float* rp = rbuf + (b0 + r) * GG + k;
        const float* mp = mem  + (b0 + r) * GG + k;
        v4f r0 = *(const v4f*)rp, r1 = *(const v4f*)(rp + 4);
        v4f m0 = *(const v4f*)mp, m1 = *(const v4f*)(mp + 4);
#pragma unroll
        for (int i = 0; i < 4; ++i) {
          out[i]     = f2bf(r0[i] * m0[i]);
          out[i + 4] = f2bf(r1[i] * m1[i]);
        }
      } else {
        out = zero8();
      }
      *(v8bf*)(At + r * 32 + c8) = out;
    }
    wait_async();
    __syncthreads();
    v16bf a = ldsA(At, lane, 0);
#pragma unroll
    for (int ti = 0; ti < 2; ++ti)
      acc[ti] = wmma_bf16(a, ldsBT(Bt, lane, (wave + ti * 8) * 16), acc[ti]);
    __syncthreads();
  }
#pragma unroll
  for (int ti = 0; ti < 2; ++ti) {
    int n = (wave + ti * 8) * 16 + (lane & 15);
    if (n < GG) {
#pragma unroll
      for (int v = 0; v < 8; ++v) {
        int b = b0 + v + ((lane & 16) ? 8 : 0);
        float h = tanhf(ahw[b * GG + n] + acc[ti][v] + bh[n] + buh[n]);
        float z = zbuf[b * GG + n];
        float mo = (1.f - z) * mem[b * GG + n] + z * h;
        mem[b * GG + n]   = mo;
        memb[b * 208 + n] = f2bf(mo);
      }
    }
  }
}

// -------- final: score = mem @ Wa + ba --------
__global__ void score_kernel(const float* __restrict__ mem, const float* __restrict__ Wa,
                             const float* __restrict__ ba, float* __restrict__ out) {
  int b = blockIdx.x * blockDim.x + threadIdx.x;
  if (b < BATCH) {
    float s = ba[0];
    for (int j = 0; j < GG; ++j) s += mem[b * GG + j] * Wa[j];
    out[b] = s;
  }
}

// ---------------- host ----------------
extern "C" void kernel_launch(void* const* d_in, const int* in_sizes, int n_in,
                              void* d_out, int out_size, void* d_ws, size_t ws_size,
                              hipStream_t stream) {
  const float* q_f  = (const float*)d_in[0];
  const float* c_f  = (const float*)d_in[1];
  const float* k_f  = (const float*)d_in[2];
  const float* Wm   = (const float*)d_in[3];
  const float* bm   = (const float*)d_in[4];
  const float* W1   = (const float*)d_in[5];
  const float* b1   = (const float*)d_in[6];
  const float* W2   = (const float*)d_in[7];
  const float* Wz   = (const float*)d_in[9];
  const float* bz   = (const float*)d_in[10];
  const float* Uz   = (const float*)d_in[11];
  const float* buz  = (const float*)d_in[12];
  const float* Wr   = (const float*)d_in[13];
  const float* br   = (const float*)d_in[14];
  const float* Ur   = (const float*)d_in[15];
  const float* bur  = (const float*)d_in[16];
  const float* Wh   = (const float*)d_in[17];
  const float* bh   = (const float*)d_in[18];
  const float* Uh   = (const float*)d_in[19];
  const float* buh  = (const float*)d_in[20];
  const float* Wa   = (const float*)d_in[21];
  const float* ba   = (const float*)d_in[22];

  char* wsb = (char*)d_ws;
  size_t off = 0;
  auto alloc = [&](size_t bytes) -> void* {
    void* p = wsb + off;
    off += (bytes + 255) & ~(size_t)255;
    return p;
  };
  float*  mem    = (float*)alloc((size_t)BATCH * GG * 4);
  __bf16* memb   = (__bf16*)alloc((size_t)BATCH * 208 * 2);
  __bf16* mib    = (__bf16*)alloc((size_t)BATCH * DD * 2);
  float*  logits = (float*)alloc((size_t)BATCH * NF * 4);
  __bf16* retb   = (__bf16*)alloc((size_t)BATCH * DD * 2);
  float*  zbuf   = (float*)alloc((size_t)BATCH * GG * 4);
  float*  rbuf   = (float*)alloc((size_t)BATCH * GG * 4);
  float*  ahw    = (float*)alloc((size_t)BATCH * GG * 4);
  __bf16* qb     = (__bf16*)alloc((size_t)BATCH * DD * 2);
  __bf16* cb     = (__bf16*)alloc((size_t)BATCH * DD * 2);
  __bf16* kb     = (__bf16*)alloc((size_t)BATCH * NF * DD * 2);
  __bf16* W1T    = (__bf16*)alloc((size_t)208 * 2400 * 2);
  __bf16* WzrhT  = (__bf16*)alloc((size_t)640 * 2816 * 2);
  __bf16* UzrT   = (__bf16*)alloc((size_t)640 * 224 * 2);
  __bf16* UhT    = (__bf16*)alloc((size_t)256 * 224 * 2);
  __bf16* WmT    = (__bf16*)alloc((size_t)512 * 224 * 2);

  dim3 pk(1024), pb(256);
  pack_bf16<<<pk, pb, 0, stream>>>(q_f, qb, BATCH * DD);
  pack_bf16<<<pk, pb, 0, stream>>>(c_f, cb, BATCH * DD);
  pack_bf16<<<pk, pb, 0, stream>>>(k_f, kb, BATCH * NF * DD);
  packT_bf16<<<pk, pb, 0, stream>>>(W1, W1T, 2400, 200, 2400, 0, 208);
  packT_bf16<<<pk, pb, 0, stream>>>(Wz, WzrhT, 2800, 200, 2816, 0,   208);
  packT_bf16<<<pk, pb, 0, stream>>>(Wr, WzrhT, 2800, 200, 2816, 208, 208);
  packT_bf16<<<pk, pb, 0, stream>>>(Wh, WzrhT, 2800, 200, 2816, 416, 208);
  packT_bf16<<<pk, pb, 0, stream>>>(Wz, WzrhT, 0, 0, 2816, 624, 16);     // zero pad cols
  packT_bf16<<<pk, pb, 0, stream>>>(Uz, UzrT, 200, 200, 224, 0,   208);
  packT_bf16<<<pk, pb, 0, stream>>>(Ur, UzrT, 200, 200, 224, 208, 208);
  packT_bf16<<<pk, pb, 0, stream>>>(Uz, UzrT, 0, 0, 224, 416, 224);      // zero pad cols
  packT_bf16<<<pk, pb, 0, stream>>>(Uh, UhT, 200, 200, 224, 0, 208);
  packT_bf16<<<pk, pb, 0, stream>>>(Uh, UhT, 0, 0, 224, 208, 48);        // zero pad cols
  packT_bf16<<<pk, pb, 0, stream>>>(Wm, WmT, 200, 400, 224, 0, 512);

  init_mem_kernel<<<(BATCH * 208 + 255) / 256, 256, 0, stream>>>(mem, memb);

  float* out_score = (float*)d_out;             // (B,1)
  float* out_att   = (float*)d_out + BATCH;     // (NHOP, B, NF)

  for (int hop = 0; hop < NHOP; ++hop) {
    mi_kernel<<<BATCH / 16, 256, 0, stream>>>(memb, WmT, bm, mib);
    logits_kernel<<<BATCH / 4, 256, 0, stream>>>(qb, cb, kb, mib, W1T, b1, W2, logits);
    softmax_retrieve_kernel<<<BATCH, 64, 0, stream>>>(
        logits, kb, out_att + (size_t)hop * BATCH * NF, retb);
    gates_kernel<<<BATCH / 16, 256, 0, stream>>>(qb, cb, retb, WzrhT, memb, UzrT,
                                                 bz, buz, br, bur, zbuf, rbuf, ahw);
    update_kernel<<<BATCH / 16, 256, 0, stream>>>(rbuf, zbuf, ahw, UhT, bh, buh, mem, memb);
  }
  score_kernel<<<BATCH / 256, 256, 0, stream>>>(mem, Wa, ba, out_score);
  (void)in_sizes; (void)n_in; (void)out_size; (void)ws_size;
}
